// BaseLUTLayer_85117661872768
// MI455X (gfx1250) — matrix operand order repaired and checked by reading, not compiled, not run on hardware
//
#include <hip/hip_runtime.h>

#define BATCH 256
#define INPUT_SIZE 8192
#define NUM_NODES 8192
#define NPAT 64
#define NPB 32            // nodes per block
#define BLOCK_BATCH 128   // batches per block (gridDim.y = 2)

typedef __attribute__((ext_vector_type(16))) _Float16     v16h;
typedef __attribute__((ext_vector_type(8)))  float        v8f;
typedef __attribute__((ext_vector_type(4)))  unsigned int v4u;
typedef __attribute__((ext_vector_type(8)))  int          v8i;
typedef __attribute__((ext_vector_type(4)))  int          v4i;

// LDS layout (dynamic shared memory, base offset 0 for the TDM target)
#define SM_RAWTAB   0                       // [32*64] f32   (8192 B)
#define SM_TAB16    8192                    // [32*64] f16   (4096 B)
#define SM_SMAP     12288                   // [32*6]  i32   ( 768 B)
#define SM_OUTTILE  13056                   // [128*32] f32  (16384 B)
#define SM_TOTAL    29440

// ---------------------------------------------------------------------------
// Kernel 1: apply flip mask and transpose x (256 x 8192) -> xT (8192 x 256).
// Makes the per-node gathers read contiguous floats instead of one element
// per 32KB-strided cacheline.
// ---------------------------------------------------------------------------
__global__ __launch_bounds__(256) void flip_transpose_kernel(
    const float* __restrict__ x,
    const unsigned char* __restrict__ fm,
    float* __restrict__ xT) {
  __shared__ float tile[32][33];
  const int c  = blockIdx.x * 32 + threadIdx.x;   // input column
  const int r0 = blockIdx.y * 32;                 // batch base
#pragma unroll
  for (int i = 0; i < 32; i += 8) {
    const int r = r0 + threadIdx.y + i;
    const size_t idx = (size_t)r * INPUT_SIZE + c;
    const float v = x[idx];
    tile[threadIdx.y + i][threadIdx.x] = fm[idx] ? 1.0f - v : v;
  }
  __syncthreads();
  const int co = blockIdx.x * 32;
#pragma unroll
  for (int i = 0; i < 32; i += 8) {
    const int row = co + threadIdx.y + i;     // output row = input column
    const int col = r0 + threadIdx.x;         // output col = batch
    xT[(size_t)row * BATCH + col] = tile[threadIdx.x][threadIdx.y + i];
  }
}

// ---------------------------------------------------------------------------
// Kernel 2: main LUT forward.
//  - TDM stages the block's 32x64 lut_table tile into LDS (tensor_load_to_lds)
//  - sigmoid applied once per node tile in LDS, packed to f16
//  - per wave: 4 nodes; per 16-batch tile: product tree (only the 16 patterns
//    this lane's WMMA A-slots need) + 2x v_wmma_f32_16x16x32_f16 with
//    column-replicated B => all D columns hold out[b, n] exactly
//  - results staged in an LDS out-tile, then written back as full 128B
//    cachelines (float4 per lane, 8 lanes per line).
// ---------------------------------------------------------------------------
__global__ __launch_bounds__(256) void lut_forward_kernel(
    const float* __restrict__ lut_table,
    const int* __restrict__ mapping,
    const float* __restrict__ xT,
    float* __restrict__ out) {
  extern __shared__ char smem[];
  float*    rawtab  = (float*)(smem + SM_RAWTAB);
  _Float16* tab16   = (_Float16*)(smem + SM_TAB16);
  int*      smap    = (int*)(smem + SM_SMAP);
  float*    outtile = (float*)(smem + SM_OUTTILE);

  const int nodeBase  = blockIdx.x * NPB;
  const int batchBase = blockIdx.y * BLOCK_BATCH;

  if (threadIdx.x < 32) {
    // ---- Tensor Data Mover: DMA 2048 contiguous f32 (32 rows x 64) into LDS.
    unsigned long long ga =
        (unsigned long long)(lut_table + (size_t)nodeBase * NPAT);
    v4u g0;
    g0[0] = 1u;                                   // count=1, user mode
    g0[1] = (unsigned int)SM_RAWTAB;              // lds_addr
    g0[2] = (unsigned int)(ga & 0xFFFFFFFFull);   // global_addr[31:0]
    g0[3] = (unsigned int)((ga >> 32) & 0x01FFFFFFull) | (2u << 30); // type=2
    v8i g1;
    g1[0] = (2 << 16);      // workgroup_mask=0, data_size=2 (4 bytes)
    g1[1] = (2048 << 16);   // tensor_dim0 = 2048 (lo16 @ bits 63:48)
    g1[2] = (1 << 16);      // tensor_dim0 hi = 0; tensor_dim1 = 1
    g1[3] = (2048 << 16);   // tensor_dim1 hi = 0; tile_dim0 = 2048
    g1[4] = 1;              // tile_dim1 = 1, tile_dim2 = 0
    g1[5] = 2048;           // tensor_dim0_stride = 2048
    g1[6] = 0;
    g1[7] = 0;
    v4i z4 = {0, 0, 0, 0};
    v8i z8 = {0, 0, 0, 0, 0, 0, 0, 0};
    __builtin_amdgcn_tensor_load_to_lds(g0, g1, z4, z4, z8, 0);
    __builtin_amdgcn_s_wait_tensorcnt(0);
  }
  if (threadIdx.x < NPB * 6) {
    smap[threadIdx.x] = mapping[(size_t)nodeBase * 6 + threadIdx.x];
  }
  __syncthreads();
#pragma unroll
  for (int i = threadIdx.x; i < NPB * NPAT; i += 256) {
    const float v = rawtab[i];
    tab16[i] = (_Float16)(1.0f / (1.0f + __expf(-v)));  // sigmoid, f16
  }
  __syncthreads();

  const int wave = threadIdx.x >> 5;
  const int lane = threadIdx.x & 31;
  const int m    = lane & 15;   // batch row within tile
  const int kg   = lane >> 4;   // half-wave K-group (== bit b3 of pattern)

  for (int j = 0; j < 4; ++j) {
    const int ni = wave * 4 + j;     // node index within block

    // B matrices: 32x16 f16, every column = sigmoid(table[n, chunk*32 + k]).
    // Lane half kg: halves 0..7 hold K=kg*8+s, halves 8..15 hold K=16+kg*8+(s-8).
    const _Float16* tb = tab16 + ni * NPAT;
    v16h B0, B1;
#pragma unroll
    for (int r = 0; r < 8; ++r) {
      const int k = kg * 8 + 2 * r + (r >= 4 ? 8 : 0);
      B0[2 * r]     = tb[k];
      B0[2 * r + 1] = tb[k + 1];
      B1[2 * r]     = tb[32 + k];
      B1[2 * r + 1] = tb[32 + k + 1];
    }
    const int i0 = smap[ni * 6 + 0], i1 = smap[ni * 6 + 1];
    const int i2 = smap[ni * 6 + 2], i3 = smap[ni * 6 + 3];
    const int i4 = smap[ni * 6 + 4], i5 = smap[ni * 6 + 5];

    for (int bt = 0; bt < BLOCK_BATCH / 16; ++bt) {
      const int b = batchBase + bt * 16 + m;
      const float g0v = xT[(size_t)i0 * BATCH + b];
      const float g1v = xT[(size_t)i1 * BATCH + b];
      const float g2v = xT[(size_t)i2 * BATCH + b];
      const float g3v = xT[(size_t)i3 * BATCH + b];
      const float g4v = xT[(size_t)i4 * BATCH + b];
      const float g5v = xT[(size_t)i5 * BATCH + b];
      const float c0 = 1.0f - g0v, c1 = 1.0f - g1v, c2 = 1.0f - g2v;
      const float c3 = 1.0f - g3v, c4 = 1.0f - g4v, c5 = 1.0f - g5v;

      // bit b3 of the pattern is fixed per half-wave by the A-matrix layout
      const float t3 = kg ? g3v : c3;

      // product tree over free bits b0..b2  (jj = b0 + 2*b1 + 4*b2)
      const float p00 = c0 * c1, p10 = g0v * c1;
      const float p01 = c0 * g1v, p11 = g0v * g1v;
      float T[8];
      T[0] = p00 * c2;  T[1] = p10 * c2;  T[2] = p01 * c2;  T[3] = p11 * c2;
      T[4] = p00 * g2v; T[5] = p10 * g2v; T[6] = p01 * g2v; T[7] = p11 * g2v;

      // fold b3 and the (b4, b5) selects into 4 scale factors
      const float u0  = t3 * c4,  u1  = t3 * g4v;
      const float m00 = u0 * c5,  m10 = u1 * c5;   // chunk 0 (b5=0), b4=0/1
      const float m01 = u0 * g5v, m11 = u1 * g5v;  // chunk 1 (b5=1), b4=0/1

      // A chunks: 16x32 f16. Lane (m,kg): halves 0..7 -> K=kg*8+jj (b4=0),
      // halves 8..15 -> K=16+kg*8+jj (b4=1); pattern = chunk*32 + K.
      v16h A0, A1;
#pragma unroll
      for (int jj = 0; jj < 8; ++jj) {
        A0[jj]     = (_Float16)(T[jj] * m00);
        A0[jj + 8] = (_Float16)(T[jj] * m10);
        A1[jj]     = (_Float16)(T[jj] * m01);
        A1[jj + 8] = (_Float16)(T[jj] * m11);
      }

      v8f acc = {};
      acc = __builtin_amdgcn_wmma_f32_16x16x32_f16(false, A0, false, B0,
                                                   (short)0, acc, false, false);
      acc = __builtin_amdgcn_wmma_f32_16x16x32_f16(false, A1, false, B1,
                                                   (short)0, acc, false, false);

      // All 16 D columns are identical (B columns replicated). Lane 0 holds
      // rows M=0..7 in acc[0..7], lane 16 holds rows M=8..15. Stage in LDS.
      if (m == 0) {
        const int lr = bt * 16 + kg * 8;   // local batch row
#pragma unroll
        for (int r = 0; r < 8; ++r) {
          outtile[(lr + r) * NPB + ni] = acc[r];
        }
      }
    }
  }
  __syncthreads();

  // Coalesced writeback: 8 lanes x float4 = one full 128B line per batch row.
  const int c4i = (threadIdx.x & 7) * 4;   // node offset within block
  const int rb  = threadIdx.x >> 3;        // starting local batch row (0..31)
  for (int r = rb; r < BLOCK_BATCH; r += 32) {
    const float4 v = *reinterpret_cast<const float4*>(outtile + r * NPB + c4i);
    *reinterpret_cast<float4*>(
        out + (size_t)(batchBase + r) * NUM_NODES + nodeBase + c4i) = v;
  }
}

// ---------------------------------------------------------------------------
extern "C" void kernel_launch(void* const* d_in, const int* in_sizes, int n_in,
                              void* d_out, int out_size, void* d_ws,
                              size_t ws_size, hipStream_t stream) {
  const float*         x       = (const float*)d_in[0];
  const float*         lut     = (const float*)d_in[1];
  const int*           mapping = (const int*)d_in[2];
  const unsigned char* fm      = (const unsigned char*)d_in[3];
  float* out = (float*)d_out;
  float* xT  = (float*)d_ws;   // 8192 * 256 * 4 = 8 MB scratch

  dim3 g1(INPUT_SIZE / 32, BATCH / 32);
  dim3 b1(32, 8);
  flip_transpose_kernel<<<g1, b1, 0, stream>>>(x, fm, xT);

  dim3 g2(NUM_NODES / NPB, BATCH / BLOCK_BATCH);
  lut_forward_kernel<<<g2, 256, SM_TOTAL, stream>>>(lut, mapping, xT, out);
}
